// GCNClassification_78735340470401
// MI455X (gfx1250) — matrix operand order repaired and checked by reading, not compiled, not run on hardware
//
#include <hip/hip_runtime.h>
#include <hip/hip_bf16.h>
#include <math.h>

typedef __attribute__((ext_vector_type(16))) __bf16 v16bf;
typedef __attribute__((ext_vector_type(8)))  __bf16 v8bf;
typedef __attribute__((ext_vector_type(4)))  __bf16 v4bf;
typedef __attribute__((ext_vector_type(8)))  float  v8f;

// ---------------- degree / norm ----------------
__global__ void k_init_deg(float* deg, int n) {
  int i = blockIdx.x * blockDim.x + threadIdx.x;
  if (i < n) deg[i] = 1.0f;                 // self-loop contributes 1
}
__global__ void k_accum_deg(const long long* __restrict__ dst, float* deg, int E) {
  int e = blockIdx.x * blockDim.x + threadIdx.x;
  if (e < E) atomicAdd(&deg[(int)dst[e]], 1.0f);
}
__global__ void k_dinv(float* deg, int n) {
  int i = blockIdx.x * blockDim.x + threadIdx.x;
  if (i < n) deg[i] = rsqrtf(deg[i]);       // deg >= 1 always
}

// ---------------- bias broadcast init ----------------
__global__ void k_init_bias(float* __restrict__ out, const float* __restrict__ b,
                            long long total, int f) {
  long long idx = (long long)blockIdx.x * blockDim.x + threadIdx.x;
  if (idx < total) out[idx] = b[(int)(idx % f)];
}

// ---------------- WMMA GEMM: out[nrows, NOUT] = A[nrows, KDIM] @ W[KDIM, ncols_real]
// (W padded to NOUT cols with zeros; optional fused ReLU on A) ----------------
template<int KDIM, int NOUT, bool RELU>
__global__ __launch_bounds__(256)
void k_gemm_wmma(const float* __restrict__ A, const float* __restrict__ W,
                 float* __restrict__ out, int nrows, int ncols_real)
{
  __shared__ __bf16 xlds[128 * 32];    // 128 rows x 32 K, bf16
  __shared__ __bf16 wlds[NOUT * 32];   // NOUT cols x 32 K (transposed), bf16

  const int tid   = threadIdx.x;
  const int lane  = tid & 31;
  const int wave  = tid >> 5;
  const int rbase = blockIdx.x * 128;

  constexpr int NT = NOUT / 16;
  v8f acc[NT];
  const v8f vzero = {0.f,0.f,0.f,0.f,0.f,0.f,0.f,0.f};
#pragma unroll
  for (int t = 0; t < NT; ++t) acc[t] = vzero;

  // CDNA5 16-bit A-matrix 16x32 layout:
  //   lanes 0-15:  M=lane,    elems 0-7 -> K=0..7,  elems 8-15 -> K=16..23
  //   lanes 16-31: M=lane-16, elems 0-7 -> K=8..15, elems 8-15 -> K=24..31
  const int arow  = lane & 15;
  const int aklo  = (lane < 16) ? 0 : 8;
  // B-matrix 32x16: lanes 0-15 hold K=0..7,16..23, lanes 16-31 hold K=8..15,24..31; N=lane&15
  const int bkoff = (lane < 16) ? 0 : 8;

  for (int kk = 0; kk < KDIM; kk += 32) {
    __syncthreads();
    // stage X tile: float4 global loads -> packed v4bf (8B) LDS stores
#pragma unroll 2
    for (int g = tid; g < 128 * 8; g += 256) {
      int row = g >> 3, k4 = (g & 7) << 2;
      int gr = rbase + row;
      float4 v = make_float4(0.f, 0.f, 0.f, 0.f);
      if (gr < nrows) v = *(const float4*)(A + (size_t)gr * KDIM + kk + k4);
      if (RELU) {
        v.x = fmaxf(v.x, 0.f); v.y = fmaxf(v.y, 0.f);
        v.z = fmaxf(v.z, 0.f); v.w = fmaxf(v.w, 0.f);
      }
      v4bf b;
      b[0] = (__bf16)v.x; b[1] = (__bf16)v.y; b[2] = (__bf16)v.z; b[3] = (__bf16)v.w;
      *(v4bf*)&xlds[(row << 5) + k4] = b;
    }
    // stage W tile transposed: wlds[n][kl] = W[(kk+kl)*NR + n], zero-pad n; 8B stores
#pragma unroll 2
    for (int g = tid; g < NOUT * 8; g += 256) {
      int n = g >> 3, k4 = (g & 7) << 2;
      v4bf b;
#pragma unroll
      for (int j = 0; j < 4; ++j) {
        float v = (n < ncols_real) ? W[(size_t)(kk + k4 + j) * ncols_real + n] : 0.f;
        b[j] = (__bf16)v;
      }
      *(v4bf*)&wlds[(n << 5) + k4] = b;
    }
    __syncthreads();

    union F { v16bf v; v8bf h[2]; };
    // A fragment (per-wave 16-row strip)
    F afrag;
    const __bf16* ap = &xlds[(wave * 16 + arow) * 32];
    afrag.h[0] = *(const v8bf*)(ap + aklo);
    afrag.h[1] = *(const v8bf*)(ap + aklo + 16);
    // load ALL B fragments first so the WMMAs can issue back-to-back
    F bfrag[NT];
#pragma unroll
    for (int t = 0; t < NT; ++t) {
      const __bf16* bp = &wlds[(t * 16 + (lane & 15)) * 32 + bkoff];
      bfrag[t].h[0] = *(const v8bf*)(bp);
      bfrag[t].h[1] = *(const v8bf*)(bp + 16);
    }
#pragma unroll
    for (int t = 0; t < NT; ++t) {
      acc[t] = __builtin_amdgcn_wmma_f32_16x16x32_bf16(
          /*neg_a=*/false, afrag.v, /*neg_b=*/false, bfrag[t].v,
          /*c_mod=*/(short)0, acc[t], /*reuse_a=*/false, /*reuse_b=*/false);
    }
  }

  // C/D layout: VGPR g, lanes 0-15 -> M=g, lanes 16-31 -> M=g+8; N=lane&15
  const int crow = (lane < 16) ? 0 : 8;
  const int ccol = lane & 15;
#pragma unroll
  for (int t = 0; t < NT; ++t) {
#pragma unroll
    for (int g = 0; g < 8; ++g) {
      int gr = rbase + wave * 16 + crow + g;
      if (gr < nrows) out[(size_t)gr * NOUT + t * 16 + ccol] = acc[t][g];
    }
  }
}

// ---------------- edge scatter: out[dst] += dinv[src]*dinv[dst] * h[src] ----------------
// grid-stride, one wave per edge; indices >= E are self-loops (i,i).
// Prefetch next iteration's source row (global_prefetch_b8) to hide gather latency.
__global__ __launch_bounds__(256)
void k_scatter(const long long* __restrict__ ei, const float* __restrict__ dinv,
               const float* __restrict__ h, float* __restrict__ out,
               int E, int n, int feats, int sstride, int dstride, int nwaves)
{
  const int lane = threadIdx.x & 31;
  const long long total = (long long)E + n;
  for (long long widx = (long long)blockIdx.x * 8 + (threadIdx.x >> 5);
       widx < total; widx += nwaves) {
    int s, d;
    if (widx < (long long)E) { s = (int)ei[widx]; d = (int)ei[(long long)E + widx]; }
    else                     { s = d = (int)(widx - E); }

    // prefetch next source row while we chew on this one
    long long nxt = widx + nwaves;
    if (nxt < total) {
      int sn = (nxt < (long long)E) ? (int)ei[nxt] : (int)(nxt - E);
      int j = lane << 2;                       // 16B apart -> covers 512B/row
      if (j < sstride)
        __builtin_prefetch(&h[(size_t)sn * sstride + j], 0, 1);
    }

    float norm = dinv[s] * dinv[d];
    const float* hs = h + (size_t)s * sstride;
    float* od = out + (size_t)d * dstride;
    for (int j = lane; j < feats; j += 32)
      atomicAdd(&od[j], norm * hs[j]);
  }
}

// ---------------- log_softmax over 33 classes, one wave per node ----------------
__global__ __launch_bounds__(256)
void k_logsoftmax(const float* __restrict__ a, float* __restrict__ out, int n)
{
  const int lane = threadIdx.x & 31;
  int node = blockIdx.x * 8 + (threadIdx.x >> 5);
  if (node >= n) return;
  const float* row = a + (size_t)node * 33;
  float v0 = row[lane];
  float v1 = (lane == 0) ? row[32] : -INFINITY;
  float m = fmaxf(v0, v1);
#pragma unroll
  for (int off = 16; off > 0; off >>= 1) m = fmaxf(m, __shfl_xor(m, off, 32));
  float e = __expf(v0 - m) + ((lane == 0) ? __expf(v1 - m) : 0.0f);
#pragma unroll
  for (int off = 16; off > 0; off >>= 1) e += __shfl_xor(e, off, 32);
  float l = __logf(e);
  out[(size_t)node * 33 + lane] = v0 - m - l;
  if (lane == 0) out[(size_t)node * 33 + 32] = row[32] - m - l;
}

// ---------------- launcher ----------------
extern "C" void kernel_launch(void* const* d_in, const int* in_sizes, int n_in,
                              void* d_out, int out_size, void* d_ws, size_t ws_size,
                              hipStream_t stream)
{
  const float*     x  = (const float*)d_in[0];
  const long long* ei = (const long long*)d_in[1];   // [2, E] int64
  const float*     W1 = (const float*)d_in[2];
  const float*     b1 = (const float*)d_in[3];
  const float*     W2 = (const float*)d_in[4];
  const float*     b2 = (const float*)d_in[5];
  float* out = (float*)d_out;

  const int N = in_sizes[0] / 256;
  const int E = in_sizes[1] / 2;

  auto al = [](size_t v) { return (v + 255) & ~(size_t)255; };
  char* p = (char*)d_ws;
  float* dinv    = (float*)p; p += al((size_t)N * 4);
  float* regionA = (float*)p; p += al((size_t)N * 128 * 4);   // h1, then h2+a2
  float* a1      = (float*)p; p += al((size_t)N * 128 * 4);
  float* h1 = regionA;
  float* h2 = regionA;                                         // reused after scatter1
  float* a2 = (float*)((char*)regionA + al((size_t)N * 48 * 4));

  const long long totE = (long long)E + N;
  const int scBlocks = 4096;                  // grid-stride scatter: 32768 waves
  const int scWaves  = scBlocks * 8;

  // normalization: deg = in-deg(dst)+1 ; dinv = deg^{-1/2}
  k_init_deg <<<(N + 255) / 256, 256, 0, stream>>>(dinv, N);
  k_accum_deg<<<(E + 255) / 256, 256, 0, stream>>>(ei + E, dinv, E);
  k_dinv     <<<(N + 255) / 256, 256, 0, stream>>>(dinv, N);

  // layer 1: h1 = X @ W1 ; a1 = scatter(norm*h1[src]->dst) + b1 ; (relu fused into GEMM2)
  k_init_bias<<<(int)(((long long)N * 128 + 255) / 256), 256, 0, stream>>>(a1, b1, (long long)N * 128, 128);
  k_gemm_wmma<256, 128, false><<<(N + 127) / 128, 256, 0, stream>>>(x, W1, h1, N, 128);
  k_scatter<<<scBlocks, 256, 0, stream>>>(ei, dinv, h1, a1, E, N, 128, 128, 128, scWaves);

  // layer 2: h2 = relu(a1) @ W2 (cols padded 33->48) ; a2 = scatter + b2
  k_gemm_wmma<128, 48, true><<<(N + 127) / 128, 256, 0, stream>>>(a1, W2, h2, N, 33);
  k_init_bias<<<(int)(((long long)N * 33 + 255) / 256), 256, 0, stream>>>(a2, b2, (long long)N * 33, 33);
  k_scatter<<<scBlocks, 256, 0, stream>>>(ei, dinv, h2, a2, E, N, 33, 48, 33, scWaves);

  // output
  k_logsoftmax<<<(N + 7) / 8, 256, 0, stream>>>(a2, out, N);
}
